// RotationScan_42606075576729
// MI455X (gfx1250) — matrix operand order repaired and checked by reading, not compiled, not run on hardware
//
#include <hip/hip_runtime.h>
#include <hip/hip_bf16.h>

// Problem constants (from reference)
constexpr int BB  = 4;
constexpr int TT  = 4096;
constexpr int DD  = 1024;
constexpr int SDD = 512;
constexpr int CHK = 32;
constexpr int NCH = TT / CHK;          // 128
constexpr int MM  = BB * TT;           // 16384
constexpr int N1  = 2 * DD;            // 2048 = SD + SD + 2SD

typedef __attribute__((ext_vector_type(16))) __bf16       bf16x16;
typedef __attribute__((ext_vector_type(8)))  float        f32x8;
typedef __attribute__((ext_vector_type(4)))  unsigned int u32x4;
typedef __attribute__((ext_vector_type(8)))  int          i32x8;
typedef __attribute__((ext_vector_type(4)))  int          i32x4;

// --- CDNA5 feature detection (device pass only) ----------------------------
#if defined(__AMDGCN__) && __has_builtin(__builtin_amdgcn_tensor_load_to_lds)
#define USE_TDM 1
#else
#define USE_TDM 0
#endif

#if USE_TDM
#if __has_builtin(__builtin_amdgcn_s_wait_tensorcnt)
#define WAIT_TENSOR0() __builtin_amdgcn_s_wait_tensorcnt(0)
#else
#define WAIT_TENSOR0() asm volatile("s_wait_tensorcnt 0x0" ::: "memory")
#endif

__device__ __forceinline__ unsigned lds_off(void* p) {
  // generic -> LDS addrspace(3) -> 32-bit LDS byte offset
  return (unsigned)(unsigned long long)(__attribute__((address_space(3))) void*)p;
}

// Issue one TDM descriptor: 2D tile of `rows` x `rowLen` bf16 elements,
// global row stride `strideElems`, into LDS at `ldsAddr` with HW padding of
// 4 DWORDs (8 bf16) after every 16 DWORDs (32 bf16) -> 40-element LDS rows.
__device__ __forceinline__ void tdm_load_tile(const unsigned short* gbase,
                                              unsigned ldsAddr,
                                              int rows, int rowLen, int strideElems) {
  unsigned long long ga = (unsigned long long)gbase;   // byte address (57-bit VA)
  u32x4 g0;
  g0[0] = 1u;                                          // count=1, flags=0
  g0[1] = ldsAddr;                                     // lds_addr (bytes)
  g0[2] = (unsigned)(ga & 0xFFFFFFFFull);              // global_addr[31:0]
  g0[3] = (unsigned)((ga >> 32) & 0x1FFFFFFull)        // global_addr[56:32]
        | (2u << 30);                                  // type=2 ("image")
  i32x8 g1;
  g1[0] = (int)((1u << 16)    // data_size = 1 -> 2 bytes
              | (1u << 20)    // pad_enable
              | (3u << 22)    // pad_interval code 3 -> every 16 DWORDs (64B row)
              | (3u << 25));  // pad_amount   code 3 -> 4 DWORDs (8 bf16)
  unsigned rl = (unsigned)rowLen, rw = (unsigned)rows;
  g1[1] = (int)((rl & 0xFFFFu) << 16);                          // tensor_dim0 lo
  g1[2] = (int)(((rl >> 16) & 0xFFFFu) | ((rw & 0xFFFFu) << 16)); // dim0 hi | dim1 lo
  g1[3] = (int)(((rw >> 16) & 0xFFFFu) | ((rl & 0xFFFFu) << 16)); // dim1 hi | tile_dim0
  g1[4] = (int)(rw & 0xFFFFu);                                  // tile_dim1 (tile_dim2=0)
  g1[5] = (int)(unsigned)strideElems;                           // tensor_dim0_stride lo32
  g1[6] = 0;                                                    // stride hi | dim1_stride lo
  g1[7] = 0;
  i32x4 z4 = {0, 0, 0, 0};                                      // 2D tensor: groups 2/3 zero
  i32x8 z8 = {0, 0, 0, 0, 0, 0, 0, 0};
  __builtin_amdgcn_tensor_load_to_lds(g0, g1, z4, z4, z8, 0);
}
#endif  // USE_TDM

__device__ __forceinline__ unsigned short f2bf(float f) {
  union { float f; unsigned u; } x; x.f = f;
  unsigned r = x.u + 0x7FFFu + ((x.u >> 16) & 1u);   // round-to-nearest-even
  return (unsigned short)(r >> 16);
}

// ---------------------------------------------------------------------------
// Elementwise f32 -> bf16 cast
// ---------------------------------------------------------------------------
__global__ void cast_f32_bf16(const float* __restrict__ src,
                              unsigned short* __restrict__ dst, int n) {
  int i = blockIdx.x * blockDim.x + threadIdx.x;
  if (i < n) dst[i] = f2bf(src[i]);
}

// ---------------------------------------------------------------------------
// Tiled transpose + cast: src f32 (R x C row-major) -> dst bf16 [n][k], n<C, k<R
// ---------------------------------------------------------------------------
__global__ void transpose_cast(const float* __restrict__ src,
                               unsigned short* __restrict__ dst,
                               int R, int C, int ldDst) {
  __shared__ float tile[32][33];
  int nT = blockIdx.x * 32, kT = blockIdx.y * 32;
  int tx = threadIdx.x, ty = threadIdx.y;       // block (32, 8)
#pragma unroll
  for (int i = 0; i < 4; ++i) {
    int k = kT + ty + i * 8;
    tile[ty + i * 8][tx] = src[(size_t)k * C + nT + tx];
  }
  __syncthreads();
#pragma unroll
  for (int i = 0; i < 4; ++i) {
    int n = nT + ty + i * 8;
    dst[(size_t)n * ldDst + kT + tx] = f2bf(tile[tx][ty + i * 8]);
  }
}

__global__ void pack_bias_k(const float* __restrict__ bt, const float* __restrict__ br,
                            const float* __restrict__ bi, float* __restrict__ bcat) {
  int i = blockIdx.x * blockDim.x + threadIdx.x;
  if (i >= N1) return;
  float v;
  if (i < 512)       v = bt[i];
  else if (i < 1024) v = br[i - 512];
  else               v = bi[i - 1024];
  bcat[i] = v;
}

// ---------------------------------------------------------------------------
// bf16 WMMA GEMM:  C[M,N] = A[M,K] * Bt[N,K]^T  (+ bias[col]) (+ resid)
// block = 256 threads (8 waves), tile 128x128, BK = 32, double-buffered LDS.
// Tile feed: TDM DMA (tensor_load_to_lds + s_wait_tensorcnt) when available,
// otherwise VGPR staging.
// ---------------------------------------------------------------------------
__launch_bounds__(256)
__global__ void gemm_bf16_wmma(const unsigned short* __restrict__ A,
                               const unsigned short* __restrict__ Bt,
                               const float* __restrict__ bias,
                               const float* __restrict__ resid,
                               float* __restrict__ C,
                               int N, int K) {
  __shared__ unsigned short sA[2][128 * 40];   // 32 K + 8 pad per row
  __shared__ unsigned short sB[2][128 * 40];

  const int tid   = threadIdx.x;
  const int lane  = tid & 31;
  const int wave  = tid >> 5;
  const int waveM = wave >> 2;                 // 0..1 -> 64-row slab
  const int waveN = wave & 3;                  // 0..3 -> 32-col slab
  const int mBase = blockIdx.y * 128;
  const int nBase = blockIdx.x * 128;

  f32x8 acc[4][2];
#pragma unroll
  for (int i = 0; i < 4; ++i)
#pragma unroll
    for (int j = 0; j < 2; ++j) acc[i][j] = (f32x8){0.f,0.f,0.f,0.f,0.f,0.f,0.f,0.f};

  const int kTiles  = K >> 5;
  const int mLane   = lane & 15;
  const int kHalfA  = (lane < 16) ? 0 : 8;     // doc 16-bit A layout
  const int kStartB = (lane < 16) ? 0 : 8;     // B layout matches A (table order)

#if !USE_TDM
  auto loadTileRegs = [&](const unsigned short* base, int rBase, int kBase, u32x4* regs) {
#pragma unroll
    for (int s = 0; s < 2; ++s) {
      int seg = tid + s * 256;
      int r   = seg >> 2;
      int so  = (seg & 3) << 3;
      regs[s] = *(const u32x4*)(base + (size_t)(rBase + r) * K + kBase + so);
    }
  };
  auto storeTileLds = [&](unsigned short* dst, const u32x4* regs) {
#pragma unroll
    for (int s = 0; s < 2; ++s) {
      int seg = tid + s * 256;
      int r   = seg >> 2;
      int so  = (seg & 3) << 3;
      *(u32x4*)(dst + r * 40 + so) = regs[s];
    }
  };
  u32x4 ra[2], rb[2];
  loadTileRegs(A,  mBase, 0, ra);
  loadTileRegs(Bt, nBase, 0, rb);
  storeTileLds(sA[0], ra);
  storeTileLds(sB[0], rb);
  __syncthreads();
#else
  if (wave == 0) {
    tdm_load_tile(A  + (size_t)mBase * K, lds_off(&sA[0][0]), 128, 32, K);
    tdm_load_tile(Bt + (size_t)nBase * K, lds_off(&sB[0][0]), 128, 32, K);
    WAIT_TENSOR0();
  }
  __syncthreads();
#endif

  for (int kt = 0; kt < kTiles; ++kt) {
    const int cur = kt & 1;
#if USE_TDM
    if (wave == 0 && kt + 1 < kTiles) {
      int kB = (kt + 1) << 5;
      tdm_load_tile(A  + (size_t)mBase * K + kB, lds_off(&sA[cur ^ 1][0]), 128, 32, K);
      tdm_load_tile(Bt + (size_t)nBase * K + kB, lds_off(&sB[cur ^ 1][0]), 128, 32, K);
    }
#else
    if (kt + 1 < kTiles) {
      int kB = (kt + 1) << 5;
      loadTileRegs(A,  mBase, kB, ra);
      loadTileRegs(Bt, nBase, kB, rb);
    }
    if (kt + 2 < kTiles) {
      int kP = (kt + 2) << 5;
      __builtin_prefetch(A  + (size_t)(mBase + (tid >> 2)) * K + kP, 0, 0);
      __builtin_prefetch(Bt + (size_t)(nBase + (tid >> 2)) * K + kP, 0, 0);
    }
#endif

    union Frag { u32x4 u[2]; bf16x16 b; };
    Frag fa[4], fb[2];
#pragma unroll
    for (int mi = 0; mi < 4; ++mi) {
      const unsigned short* p = &sA[cur][(waveM * 64 + mi * 16 + mLane) * 40 + kHalfA];
      fa[mi].u[0] = *(const u32x4*)(p);
      fa[mi].u[1] = *(const u32x4*)(p + 16);
    }
#pragma unroll
    for (int ni = 0; ni < 2; ++ni) {
      const unsigned short* p = &sB[cur][(waveN * 32 + ni * 16 + mLane) * 40 + kStartB];
      fb[ni].u[0] = *(const u32x4*)(p);
      fb[ni].u[1] = *(const u32x4*)(p + 16);
    }
#pragma unroll
    for (int mi = 0; mi < 4; ++mi)
#pragma unroll
      for (int ni = 0; ni < 2; ++ni)
        acc[mi][ni] = __builtin_amdgcn_wmma_f32_16x16x32_bf16(
            false, fa[mi].b, false, fb[ni].b, (short)0, acc[mi][ni], false, false);

#if USE_TDM
    if (wave == 0 && kt + 1 < kTiles) WAIT_TENSOR0();
    __syncthreads();
#else
    __syncthreads();
    if (kt + 1 < kTiles) {
      storeTileLds(sA[(kt + 1) & 1], ra);
      storeTileLds(sB[(kt + 1) & 1], rb);
    }
    __syncthreads();
#endif
  }

  // Epilogue: C/D layout — VGPR v holds M = v (lanes 0-15) / v+8 (lanes 16-31)
  const int rOff = (lane < 16) ? 0 : 8;
#pragma unroll
  for (int mi = 0; mi < 4; ++mi)
#pragma unroll
    for (int ni = 0; ni < 2; ++ni) {
      int col = nBase + waveN * 32 + ni * 16 + mLane;
#pragma unroll
      for (int v = 0; v < 8; ++v) {
        int row = mBase + waveM * 64 + mi * 16 + v + rOff;
        float val = acc[mi][ni][v] + bias[col];
        if (resid) val += resid[(size_t)row * N + col];
        C[(size_t)row * N + col] = val;
      }
    }
}

// ---------------------------------------------------------------------------
// Scan phase 1: per-chunk affine summary (A_c, B_c) of h_t = a_t h + b_t
// ---------------------------------------------------------------------------
__global__ void scan_chunk_reduce(const float* __restrict__ act,
                                  float* __restrict__ Ar, float* __restrict__ Ai,
                                  float* __restrict__ Br, float* __restrict__ Bi) {
  int idx = blockIdx.x * blockDim.x + threadIdx.x;     // B*NCH*SD
  int d = idx % SDD;
  int c = (idx / SDD) % NCH;
  int b = idx / (SDD * NCH);
  float ar = 1.f, ai = 0.f, br_ = 0.f, bi_ = 0.f;
  size_t base0 = ((size_t)(b * TT + c * CHK)) * (size_t)N1 + d;
#pragma unroll 4
  for (int j = 0; j < CHK; ++j) {
    size_t base = base0 + (size_t)j * N1;
    float th = act[base];
    float rp = act[base + 512];
    float ir = act[base + 1024];
    float ii = act[base + 1536];
    float ret = 1.f / (1.f + __expf(-rp));
    float mag = fmaxf(ret, 1e-6f);
    float s, cs; __sincosf(th, &s, &cs);
    float car = mag * cs, cai = mag * s;
    float drv = 1.f - ret;
    float nbr = car * br_ - cai * bi_ + drv * ir;
    float nbi = car * bi_ + cai * br_ + drv * ii;
    br_ = nbr; bi_ = nbi;
    float nar = car * ar - cai * ai;
    float nai = car * ai + cai * ar;
    ar = nar; ai = nai;
  }
  size_t ci = ((size_t)b * NCH + c) * SDD + d;
  Ar[ci] = ar; Ai[ci] = ai; Br[ci] = br_; Bi[ci] = bi_;
}

// ---------------------------------------------------------------------------
// Scan phase 2: sequential carry over 128 chunks (2048 independent lanes)
// ---------------------------------------------------------------------------
__global__ void scan_carry(const float* __restrict__ Ar, const float* __restrict__ Ai,
                           const float* __restrict__ Br, const float* __restrict__ Bi,
                           float* __restrict__ Hr, float* __restrict__ Hi) {
  int idx = blockIdx.x * blockDim.x + threadIdx.x;
  if (idx >= BB * SDD) return;
  int d = idx % SDD, b = idx / SDD;
  float hr = 0.f, hi = 0.f;
  for (int c = 0; c < NCH; ++c) {
    size_t i = ((size_t)b * NCH + c) * SDD + d;
    Hr[i] = hr; Hi[i] = hi;                    // h entering this chunk
    float ar = Ar[i], ai = Ai[i];
    float nr = ar * hr - ai * hi + Br[i];
    float ni = ar * hi + ai * hr + Bi[i];
    hr = nr; hi = ni;
  }
}

// ---------------------------------------------------------------------------
// Scan phase 3: apply — recompute within chunk, emit bf16 [out_r | out_i]
// ---------------------------------------------------------------------------
__global__ void scan_apply(const float* __restrict__ act,
                           const float* __restrict__ Hr, const float* __restrict__ Hi,
                           unsigned short* __restrict__ outbf) {
  int idx = blockIdx.x * blockDim.x + threadIdx.x;
  int d = idx % SDD;
  int c = (idx / SDD) % NCH;
  int b = idx / (SDD * NCH);
  size_t ci = ((size_t)b * NCH + c) * SDD + d;
  float hr = Hr[ci], hi = Hi[ci];
  size_t base0 = ((size_t)(b * TT + c * CHK)) * (size_t)N1 + d;
#pragma unroll 4
  for (int j = 0; j < CHK; ++j) {
    size_t base = base0 + (size_t)j * N1;
    float th = act[base];
    float rp = act[base + 512];
    float ir = act[base + 1024];
    float ii = act[base + 1536];
    float ret = 1.f / (1.f + __expf(-rp));
    float mag = fmaxf(ret, 1e-6f);
    float s, cs; __sincosf(th, &s, &cs);
    float car = mag * cs, cai = mag * s;
    float drv = 1.f - ret;
    float nr = car * hr - cai * hi + drv * ir;
    float ni = car * hi + cai * hr + drv * ii;
    hr = nr; hi = ni;
    size_t row = (size_t)(b * TT + c * CHK + j);
    outbf[row * DD + d]       = f2bf(hr);
    outbf[row * DD + 512 + d] = f2bf(hi);
  }
}

// ---------------------------------------------------------------------------
// Row LayerNorm (in-place on y = d_out), D = 1024, 256 threads/row
// ---------------------------------------------------------------------------
__launch_bounds__(256)
__global__ void layernorm_k(float* __restrict__ y,
                            const float* __restrict__ gamma,
                            const float* __restrict__ beta) {
  int row = blockIdx.x, tid = threadIdx.x;
  float* p = y + (size_t)row * DD;
  float vals[4];
  float s = 0.f, s2 = 0.f;
#pragma unroll
  for (int i = 0; i < 4; ++i) {
    float v = p[tid + i * 256];
    vals[i] = v; s += v; s2 += v * v;
  }
  __shared__ float red[256], red2[256];
  red[tid] = s; red2[tid] = s2;
  __syncthreads();
  for (int o = 128; o > 0; o >>= 1) {
    if (tid < o) { red[tid] += red[tid + o]; red2[tid] += red2[tid + o]; }
    __syncthreads();
  }
  float mu  = red[0] * (1.f / DD);
  float var = red2[0] * (1.f / DD) - mu * mu;
  float inv = rsqrtf(var + 1e-5f);
#pragma unroll
  for (int i = 0; i < 4; ++i) {
    int cidx = tid + i * 256;
    p[cidx] = (vals[i] - mu) * inv * gamma[cidx] + beta[cidx];
  }
}

// ---------------------------------------------------------------------------
// Workspace layout (bytes)
// ---------------------------------------------------------------------------
constexpr size_t OFF_XBF   = 0;                                     // 16384*1024*2
constexpr size_t OFF_WCT   = OFF_XBF + (size_t)MM * DD * 2;         // 2048*1024*2
constexpr size_t OFF_BCT   = OFF_WCT + (size_t)N1 * DD * 2;         // 2048*4
constexpr size_t OFF_WOT   = OFF_BCT + (size_t)N1 * 4;              // 1024*1024*2
constexpr size_t OFF_ACT   = OFF_WOT + (size_t)DD * DD * 2;         // 16384*2048*4
constexpr size_t SZ_CH     = (size_t)BB * NCH * SDD * 4;            // 1 MiB each
constexpr size_t OFF_AR    = OFF_ACT + (size_t)MM * N1 * 4;
constexpr size_t OFF_AI    = OFF_AR + SZ_CH;
constexpr size_t OFF_BR    = OFF_AI + SZ_CH;
constexpr size_t OFF_BI    = OFF_BR + SZ_CH;
constexpr size_t OFF_HR    = OFF_BI + SZ_CH;
constexpr size_t OFF_HI    = OFF_HR + SZ_CH;
constexpr size_t OFF_OUTBF = OFF_HI + SZ_CH;                        // 16384*1024*2

extern "C" void kernel_launch(void* const* d_in, const int* in_sizes, int n_in,
                              void* d_out, int out_size, void* d_ws, size_t ws_size,
                              hipStream_t stream) {
  (void)in_sizes; (void)n_in; (void)out_size; (void)ws_size;

  const float* x     = (const float*)d_in[0];
  const float* Wt    = (const float*)d_in[1];
  const float* bt    = (const float*)d_in[2];
  const float* Wr    = (const float*)d_in[3];
  const float* br    = (const float*)d_in[4];
  const float* Wi    = (const float*)d_in[5];
  const float* bi    = (const float*)d_in[6];
  const float* Wo    = (const float*)d_in[7];
  const float* bo    = (const float*)d_in[8];
  const float* gamma = (const float*)d_in[9];
  const float* beta  = (const float*)d_in[10];
  float* out = (float*)d_out;

  char* ws = (char*)d_ws;
  unsigned short* xbf   = (unsigned short*)(ws + OFF_XBF);
  unsigned short* WcatT = (unsigned short*)(ws + OFF_WCT);
  float*          bcat  = (float*)(ws + OFF_BCT);
  unsigned short* WoT   = (unsigned short*)(ws + OFF_WOT);
  float*          act   = (float*)(ws + OFF_ACT);
  float* Ar = (float*)(ws + OFF_AR);
  float* Ai = (float*)(ws + OFF_AI);
  float* Br = (float*)(ws + OFF_BR);
  float* Bi = (float*)(ws + OFF_BI);
  float* Hr = (float*)(ws + OFF_HR);
  float* Hi = (float*)(ws + OFF_HI);
  unsigned short* outbf = (unsigned short*)(ws + OFF_OUTBF);

  // 1) x -> bf16
  {
    int n = MM * DD;
    cast_f32_bf16<<<n / 256, 256, 0, stream>>>(x, xbf, n);
  }
  // 2) Pre-transposed bf16 weights: WcatT[n][k] over [Wt|Wr|Wi], WoT[n][k]
  {
    dim3 blk(32, 8);
    transpose_cast<<<dim3(512 / 32, DD / 32), blk, 0, stream>>>(Wt, WcatT, DD, 512, DD);
    transpose_cast<<<dim3(512 / 32, DD / 32), blk, 0, stream>>>(Wr, WcatT + (size_t)512 * DD, DD, 512, DD);
    transpose_cast<<<dim3(1024 / 32, DD / 32), blk, 0, stream>>>(Wi, WcatT + (size_t)1024 * DD, DD, 1024, DD);
    transpose_cast<<<dim3(1024 / 32, DD / 32), blk, 0, stream>>>(Wo, WoT, DD, 1024, DD);
    pack_bias_k<<<N1 / 256, 256, 0, stream>>>(bt, br, bi, bcat);
  }
  // 3) GEMM1: act[M, 2048] = x @ [Wt|Wr|Wi] + bcat
  gemm_bf16_wmma<<<dim3(N1 / 128, MM / 128), 256, 0, stream>>>(
      xbf, WcatT, bcat, nullptr, act, N1, DD);
  // 4) Rotation scan: chunk reduce -> carry -> apply (writes bf16 out)
  {
    int total = BB * NCH * SDD;                 // 262144
    scan_chunk_reduce<<<total / 256, 256, 0, stream>>>(act, Ar, Ai, Br, Bi);
    scan_carry<<<(BB * SDD) / 256, 256, 0, stream>>>(Ar, Ai, Br, Bi, Hr, Hi);
    scan_apply<<<total / 256, 256, 0, stream>>>(act, Hr, Hi, outbf);
  }
  // 5) GEMM2: out = x + scan_out @ Wo + bo
  gemm_bf16_wmma<<<dim3(DD / 128, MM / 128), 256, 0, stream>>>(
      outbf, WoT, bo, x, out, DD, DD);
  // 6) LayerNorm in-place on d_out
  layernorm_k<<<MM, 256, 0, stream>>>(out, gamma, beta);
}
